// SemanticAwareAttention_Mask_30666066494067
// MI455X (gfx1250) — compile-verified, matrix operand order checked by reading, not run on hardware
//
#include <hip/hip_runtime.h>
#include <math.h>

// ---------------- problem constants ----------------
#define BB 8
#define NN 8192
#define CC 20
#define EE 512
#define TM 128          // rows per block tile
#define NTILES (NN/TM)  // 64
#define KT16 16         // number of K-tiles of 32 in E=512

// ---------------- workspace layout (bytes) ----------------
#define WQB_OFF 0                          // Wq bf16, B-frag swizzled: 512*512*2 = 524288
#define KPB_OFF 524288                     // kp^T bf16 B-frags: 8*2*16*512*2 = 262144
#define VPB_OFF 786432                     // vp bf16 B-frags (K=32 padded): 8*32*512*2 = 262144
#define WS_TOTAL 1048576

// ---------------- main-kernel LDS layout (bytes) ----------------
#define A_OFF   0        // q A-frags bf16: 128 KB
#define QP_OFF  131072   // qp A-frags bf16: 128 KB
#define AT_OFF  262144   // attn logits f32 [128][32]: 16 KB
#define A2_OFF  278528   // attn2 A-frag bf16 [8][32][16]: 8 KB
#define BS_OFF  286720   // Wq B-tile double-buffer stage: 2 x 16 KB
#define BS_SZ   16384
#define SMEM_MAIN 319488
#define SMEM_KV   81920  // prep_kv: 2*20*512 f32

#ifndef __has_builtin
#define __has_builtin(x) 0
#endif
#if defined(__HIP_DEVICE_COMPILE__) && \
    __has_builtin(__builtin_amdgcn_global_load_async_to_lds_b128) && \
    __has_builtin(__builtin_amdgcn_s_wait_asynccnt)
#define USE_ASYNC_LDS 1
#else
#define USE_ASYNC_LDS 0
#endif

typedef __bf16 v16bf __attribute__((ext_vector_type(16)));
typedef float  v8f   __attribute__((ext_vector_type(8)));
typedef int    v4i   __attribute__((ext_vector_type(4)));
typedef __attribute__((address_space(1))) v4i* gv4i_p;   // global (__device__) int4*
typedef __attribute__((address_space(3))) v4i* lv4i_p;   // LDS int4*

static __device__ __forceinline__ unsigned short f2bf(float f) {
  union { float f; unsigned u; } x; x.f = f;
  unsigned r = x.u + 0x7fffu + ((x.u >> 16) & 1u);   // RNE
  return (unsigned short)(r >> 16);
}
static __device__ __forceinline__ unsigned pack2(float lo, float hi) {
  return (unsigned)f2bf(lo) | ((unsigned)f2bf(hi) << 16);
}
// 16-bit WMMA A/B fragment swizzle: element (base16 = row/col within 16, kk in 0..31)
static __device__ __forceinline__ int frag_lane(int base16, int kk) { return base16 + 16 * ((kk >> 3) & 1); }
static __device__ __forceinline__ int frag_idx (int kk)             { return (kk & 7) + 8 * ((kk >> 4) & 1); }

static __device__ __forceinline__ v8f wmma_bf16(v16bf a, v16bf b, v8f c) {
  return __builtin_amdgcn_wmma_f32_16x16x32_bf16(false, a, false, b, (short)0, c, false, false);
}

#if USE_ASYNC_LDS
// Async global->LDS copy of one 16 KB Wq stage (ASYNCcnt-tracked, no VGPR data path).
// Dynamic LDS starts at offset 0 (no static LDS in this kernel), so the LDS
// destination can be formed directly from the byte offset.
static __device__ __forceinline__ void stage_async16k(const unsigned short* wqb, int et,
                                                      int ldsByteOff, int tid) {
  const char* g = (const char*)(wqb + (et * 16) * 512) + tid * 16;
#pragma unroll
  for (int i = 0; i < 4; ++i) {
    __builtin_amdgcn_global_load_async_to_lds_b128(
        (gv4i_p)(g + i * 4096),
        (lv4i_p)(unsigned long)(ldsByteOff + tid * 16 + i * 4096),
        0, 0);
  }
}
#endif

// ---------------- prep: zero kp/vp B-frag regions (class padding 20->32) ----------------
__global__ void saam_zero_pad(unsigned int* w) {
  unsigned i = blockIdx.x * 256u + threadIdx.x;     // 512 blocks * 256 = 131072 words = 512 KB
  w[i] = 0u;
}

// ---------------- prep: Wq f32 -> bf16 B-fragment swizzle ----------------
__global__ void saam_prep_wq(const float* __restrict__ Wq, unsigned short* __restrict__ wqb) {
  int id = blockIdx.x * 256 + threadIdx.x;          // 1024 blocks -> 262144 elements
  int k = id >> 9, e = id & 511;
  int et = e >> 4, n16 = e & 15;
  int kt = k >> 5, kk = k & 31;
  int lane = frag_lane(n16, kk), idx = frag_idx(kk);
  wqb[((et * 16 + kt) * 32 + lane) * 16 + idx] = f2bf(Wq[id]);
}

// ---------------- prep: kp = k@Wk+bk, vp = v@Wv+bv, stored as swizzled bf16 B-frags ----
__global__ void saam_prep_kv(const float* __restrict__ kin, const float* __restrict__ vin,
                             const float* __restrict__ Wk, const float* __restrict__ bk,
                             const float* __restrict__ Wv, const float* __restrict__ bv,
                             unsigned short* __restrict__ kpb, unsigned short* __restrict__ vpb) {
  extern __shared__ char pk_smem[];
  float* rowk = (float*)pk_smem;          // [20][512]
  float* rowv = rowk + CC * EE;           // [20][512]
  int b = blockIdx.x >> 2, eq = blockIdx.x & 3;
  int tid = threadIdx.x;
  for (int i = tid; i < CC * EE; i += 128) {
    rowk[i] = kin[b * CC * EE + i];
    rowv[i] = vin[b * CC * EE + i];
  }
  __syncthreads();
  int e = eq * 128 + tid;
  float ka[CC], va[CC];
  float bke = bk[e], bve = bv[e];
#pragma unroll
  for (int c = 0; c < CC; ++c) { ka[c] = bke; va[c] = bve; }
  for (int d = 0; d < EE; ++d) {
    float wk = Wk[d * EE + e];
    float wv = Wv[d * EE + e];
#pragma unroll
    for (int c = 0; c < CC; ++c) {
      ka[c] += rowk[c * EE + d] * wk;
      va[c] += rowv[c * EE + d] * wv;
    }
  }
#pragma unroll
  for (int c = 0; c < CC; ++c) {
    { // kp^T B-frag: K-dim = e (512), N-dim = c (pad 32)
      int nt = c >> 4, n16 = c & 15, kt = e >> 5, kk = e & 31;
      kpb[(((b * 2 + nt) * 16 + kt) * 32 + frag_lane(n16, kk)) * 16 + frag_idx(kk)] = f2bf(ka[c]);
    }
    { // vp B-frag: K-dim = c (pad 32), N-dim = e (512)
      int et = e >> 4, n16 = e & 15, kk = c;
      vpb[((b * 32 + et) * 32 + frag_lane(n16, kk)) * 16 + frag_idx(kk)] = f2bf(va[c]);
    }
  }
}

// ---------------- fused main kernel ----------------
__global__ __launch_bounds__(256, 1)
void saam_main(const float* __restrict__ q, const float* __restrict__ mask,
               const float* __restrict__ cpred, const float* __restrict__ bq,
               const unsigned short* __restrict__ wqb, const unsigned short* __restrict__ kpb,
               const unsigned short* __restrict__ vpb, float* __restrict__ out) {
  extern __shared__ char smem[];
  unsigned short* aL  = (unsigned short*)(smem + A_OFF);
  unsigned short* qpL = (unsigned short*)(smem + QP_OFF);
  float*          atL = (float*)(smem + AT_OFF);
  unsigned short* a2L = (unsigned short*)(smem + A2_OFF);

  const int tid = threadIdx.x;
  const int w = tid >> 5, l = tid & 31;     // wave id (owns M-tile), lane
  const int b  = blockIdx.x / NTILES;
  const int n0 = (blockIdx.x % NTILES) * TM;
  const float* qrow = q + (size_t)b * NN * EE + (size_t)n0 * EE;

  // ---- P0: load q tile (f32), convert bf16, store as A-fragments into LDS ----
#pragma unroll 4
  for (int r = 0; r < 32; ++r) {
    int run = r * 256 + tid;            // 128 rows * 64 groups of 8 k-values
    int m = run >> 6, kg = run & 63;
    const float4* src = (const float4*)(qrow + m * EE + kg * 8);
    float4 x0 = src[0], x1 = src[1];
    uint4 pkd = make_uint4(pack2(x0.x, x0.y), pack2(x0.z, x0.w),
                           pack2(x1.x, x1.y), pack2(x1.z, x1.w));
    int mt = m >> 4, m16 = m & 15;
    int kt = kg >> 2;
    int half = kg & 1;                  // (kk0>>3)&1
    int ib = (kg >> 1) & 1;             // (kk0>>4)&1
    *(uint4*)(aL + (((mt * 16 + kt) * 32) + (m16 + 16 * half)) * 16 + ib * 8) = pkd;
  }
  { // zero attn2 A-fragment region (class padding)
    unsigned int* a2w = (unsigned int*)a2L;
    for (int i = tid; i < 2048; i += 256) a2w[i] = 0u;
  }

  // ---- P1: qp = q @ Wq + bq, double-buffered Wq staging pipelined under WMMA ----
  const int n16 = l & 15;
  const int mrow = (l >> 4) * 8;
#if USE_ASYNC_LDS
  stage_async16k(wqb, 0, BS_OFF, tid);
  __builtin_amdgcn_s_wait_asynccnt(0);
  __syncthreads();
#else
  uint4 pf[4];
  {
    const uint4* g = (const uint4*)wqb;
#pragma unroll
    for (int i = 0; i < 4; ++i) pf[i] = g[tid + i * 256];
    uint4* d = (uint4*)(smem + BS_OFF);
#pragma unroll
    for (int i = 0; i < 4; ++i) d[tid + i * 256] = pf[i];
  }
  __syncthreads();
#endif
  for (int et = 0; et < 32; ++et) {
    const int cur = et & 1, nxt = cur ^ 1;
    // issue next stage's global traffic before touching the WMMA pipe
#if USE_ASYNC_LDS
    if (et < 31) stage_async16k(wqb, et + 1, BS_OFF + nxt * BS_SZ, tid);
#else
    if (et < 31) {
      const uint4* g = (const uint4*)(wqb + ((et + 1) * 16) * 512);
#pragma unroll
      for (int i = 0; i < 4; ++i) pf[i] = g[tid + i * 256];
    }
#endif
    const unsigned short* bsC = (const unsigned short*)(smem + BS_OFF + cur * BS_SZ);
    v8f acc = (v8f){};
    for (int kt = 0; kt < KT16; ++kt) {
      v16bf a  = *(const v16bf*)(aL  + ((w * 16 + kt) * 32 + l) * 16);
      v16bf bb = *(const v16bf*)(bsC + (kt * 32 + l) * 16);
      acc = wmma_bf16(a, bb, acc);
    }
    // epilogue: +bq, scatter qp into A-frag layout for the attention GEMM
    {
      float bqv = bq[et * 16 + n16];
      int kt_q = et >> 1;
      int kk = (et & 1) * 16 + n16;
      int half = (kk >> 3) & 1;
      int idx = frag_idx(kk);
#pragma unroll
      for (int r = 0; r < 8; ++r) {
        qpL[((w * 16 + kt_q) * 32 + (mrow + r + 16 * half)) * 16 + idx] = f2bf(acc[r] + bqv);
      }
    }
    if (et < 31) {
#if USE_ASYNC_LDS
      __builtin_amdgcn_s_wait_asynccnt(0);
#else
      uint4* d = (uint4*)(smem + BS_OFF + nxt * BS_SZ);
#pragma unroll
      for (int i = 0; i < 4; ++i) d[tid + i * 256] = pf[i];
#endif
      __syncthreads();
    }
  }
  __syncthreads();

  // ---- P2: attn = qp @ kp^T  (N = 32 padded classes, K = 512) ----
  {
    v8f c0 = (v8f){}; v8f c1 = (v8f){};
    for (int kt = 0; kt < KT16; ++kt) {
      v16bf a  = *(const v16bf*)(qpL + ((w * 16 + kt) * 32 + l) * 16);
      v16bf b0 = *(const v16bf*)(kpb + (((b * 2 + 0) * 16 + kt) * 32 + l) * 16);
      v16bf b1 = *(const v16bf*)(kpb + (((b * 2 + 1) * 16 + kt) * 32 + l) * 16);
      c0 = wmma_bf16(a, b0, c0);
      c1 = wmma_bf16(a, b1, c1);
    }
#pragma unroll
    for (int r = 0; r < 8; ++r) {
      int m = w * 16 + mrow + r;
      atL[m * 32 + n16]      = c0[r];
      atL[m * 32 + 16 + n16] = c1[r];
    }
  }
  __syncthreads();

  // ---- P3: per-row double softmax (+ coarse_pred softmax), write attn2 A-frag ----
  if (tid < TM) {
    const float scale = 0.044194173824159216f;   // 1/sqrt(512)
    int m = tid;
    float lg[CC], mk[CC], cp[CC];
#pragma unroll
    for (int c = 0; c < CC; ++c) mk[c] = mask[b * CC + c];
#pragma unroll
    for (int c = 0; c < CC; ++c) lg[c] = atL[m * 32 + c] * scale + mk[c];
    float mx = -1e30f, s;
#pragma unroll
    for (int c = 0; c < CC; ++c) mx = fmaxf(mx, lg[c]);
    s = 0.f;
#pragma unroll
    for (int c = 0; c < CC; ++c) { lg[c] = __expf(lg[c] - mx); s += lg[c]; }
    float inv = __frcp_rn(s);
#pragma unroll
    for (int c = 0; c < CC; ++c) lg[c] *= inv;
    // coarse_pred softmax over classes (transposed read)
#pragma unroll
    for (int c = 0; c < CC; ++c) cp[c] = cpred[(size_t)(b * CC + c) * NN + n0 + m];
    mx = -1e30f;
#pragma unroll
    for (int c = 0; c < CC; ++c) mx = fmaxf(mx, cp[c]);
    s = 0.f;
#pragma unroll
    for (int c = 0; c < CC; ++c) { cp[c] = __expf(cp[c] - mx); s += cp[c]; }
    inv = __frcp_rn(s);
#pragma unroll
    for (int c = 0; c < CC; ++c) cp[c] *= inv;
    // second softmax: softmax(attn*cp + mask)
#pragma unroll
    for (int c = 0; c < CC; ++c) lg[c] = lg[c] * cp[c] + mk[c];
    mx = -1e30f;
#pragma unroll
    for (int c = 0; c < CC; ++c) mx = fmaxf(mx, lg[c]);
    s = 0.f;
#pragma unroll
    for (int c = 0; c < CC; ++c) { lg[c] = __expf(lg[c] - mx); s += lg[c]; }
    inv = __frcp_rn(s);
    int mt = m >> 4, m16 = m & 15;
#pragma unroll
    for (int c = 0; c < CC; ++c) {
      a2L[(mt * 32 + frag_lane(m16, c)) * 16 + frag_idx(c)] = f2bf(lg[c] * inv);
    }
  }
  __syncthreads();

  // ---- P4: out = attn2 @ vp  (K = 32 padded, one WMMA per 16x16 tile) ----
  {
    v16bf a2 = *(const v16bf*)(a2L + (w * 32 + l) * 16);
    float* outp = out + (size_t)b * NN * EE + (size_t)n0 * EE;
    for (int et = 0; et < 32; ++et) {
      v16bf bv = *(const v16bf*)(vpb + ((b * 32 + et) * 32 + l) * 16);
      v8f oc = (v8f){};
      oc = wmma_bf16(a2, bv, oc);
#pragma unroll
      for (int r = 0; r < 8; ++r) {
        int m = w * 16 + mrow + r;
        outp[(size_t)m * EE + et * 16 + n16] = oc[r];
      }
    }
  }
}

// ---------------- host launcher ----------------
extern "C" void kernel_launch(void* const* d_in, const int* in_sizes, int n_in,
                              void* d_out, int out_size, void* d_ws, size_t ws_size,
                              hipStream_t stream) {
  (void)in_sizes; (void)n_in; (void)out_size; (void)ws_size;
  const float* q     = (const float*)d_in[0];
  const float* k     = (const float*)d_in[1];
  const float* v     = (const float*)d_in[2];
  const float* mask  = (const float*)d_in[3];
  const float* cpred = (const float*)d_in[4];
  const float* Wq    = (const float*)d_in[5];
  const float* bq    = (const float*)d_in[6];
  const float* Wk    = (const float*)d_in[7];
  const float* bk    = (const float*)d_in[8];
  const float* Wv    = (const float*)d_in[9];
  const float* bv    = (const float*)d_in[10];
  float* out = (float*)d_out;

  unsigned short* wqb = (unsigned short*)((char*)d_ws + WQB_OFF);
  unsigned short* kpb = (unsigned short*)((char*)d_ws + KPB_OFF);
  unsigned short* vpb = (unsigned short*)((char*)d_ws + VPB_OFF);

  (void)hipFuncSetAttribute(reinterpret_cast<const void*>(saam_main),
                            hipFuncAttributeMaxDynamicSharedMemorySize, SMEM_MAIN);
  (void)hipFuncSetAttribute(reinterpret_cast<const void*>(saam_prep_kv),
                            hipFuncAttributeMaxDynamicSharedMemorySize, SMEM_KV);

  // zero the padded kp/vp fragment regions (512 KB)
  saam_zero_pad<<<512, 256, 0, stream>>>((unsigned int*)((char*)d_ws + KPB_OFF));
  // Wq -> swizzled bf16 B-fragments
  saam_prep_wq<<<1024, 256, 0, stream>>>(Wq, wqb);
  // kp/vp projections -> swizzled bf16 B-fragments
  saam_prep_kv<<<BB * 4, 128, SMEM_KV, stream>>>(k, v, Wk, bk, Wv, bv, kpb, vpb);
  // fused attention
  saam_main<<<BB * NTILES, 256, SMEM_MAIN, stream>>>(q, mask, cpred, bq, wqb, kpb, vpb, out);
}
